// Graph2dConvolution_55224689492271
// MI455X (gfx1250) — compile-verified, matrix operand order checked by reading, not run on hardware
//
#include <hip/hip_runtime.h>
#include <stdint.h>

#define HW    65536
#define CIN   128
#define COUT  256
#define KSEG  64
#define LDA   136   // 128 + 8 pad (u16 elements) to break LDS bank conflicts

typedef __bf16 bf16_t;
typedef bf16_t v16bf __attribute__((ext_vector_type(16)));
typedef float  v8f   __attribute__((ext_vector_type(8)));
typedef unsigned short u16;
typedef u16 v8u16  __attribute__((ext_vector_type(8)));
typedef u16 v16u16 __attribute__((ext_vector_type(16)));
typedef float v4f  __attribute__((ext_vector_type(4)));

__device__ __forceinline__ u16 f2bf(float f) {
  uint32_t u = __builtin_bit_cast(uint32_t, f);
  u += 0x7FFFu + ((u >> 16) & 1u);   // round-to-nearest-even
  return (u16)(u >> 16);
}

// Load a 16x32 bf16 operand fragment per the CDNA5 ISA layout:
// lane[3:0] selects the row, lane[4] selects K-halves {0..7,16..23} vs {8..15,24..31}.
// Rows are stored K-contiguous in LDS, so each lane reads two aligned 16B chunks.
__device__ __forceinline__ v16bf lds_frag(const u16* p) {
  v8u16 lo = *(const v8u16*)(p);
  v8u16 hi = *(const v8u16*)(p + 16);
  v16u16 cat = __builtin_shufflevector(lo, hi,
      0,1,2,3,4,5,6,7,8,9,10,11,12,13,14,15);
  return __builtin_bit_cast(v16bf, cat);
}

// Fused 1x1-conv GEMM:  xp[b, d0+0..127, pix0+0..511] = weight^T * x
// SEG=true : accumulate per-segment sums (LDS reduce -> global atomics), no store.
// SEG=false: recompute xp, add broadcast s[b,d], store to out.
template<bool SEG>
__global__ __launch_bounds__(256) void conv_gemm_kernel(
    const float* __restrict__ x, const int* __restrict__ index,
    const float* __restrict__ weight, const float* __restrict__ s,
    float* __restrict__ out, float* __restrict__ seg_sum,
    float* __restrict__ cnt)
{
  __shared__ u16   Ash[128 * LDA];      // A^T tile: [m=Cout_local][k=Cin] bf16
  __shared__ u16   Bsh[128 * LDA];      // x tile transposed: [n=pix_local][k=Cin] bf16
  __shared__ float segsh[KSEG * 128];   // per-block segment partial sums
  __shared__ float cntsh[KSEG];
  __shared__ int   idxsh[128];
  __shared__ float ssh[128];

  const int tid   = threadIdx.x;
  const int lane  = tid & 31;
  const int wave  = tid >> 5;
  const int m_off = (wave & 3) * 32;    // 4 waves tile M (32 rows each)
  const int n_off = (wave >> 2) * 64;   // 2 waves tile N (64 cols each)
  const int b      = blockIdx.z;
  const int m_tile = blockIdx.y;        // 0..1  (Cout halves)
  const int d0     = m_tile * 128;
  const int pix0   = blockIdx.x * 512;

  // Stage A tile once: At[m][k] = weight[k][d0+m] (coalesced over m).
  for (int i = 0; i < 64; ++i) {
    int v = tid + i * 256;
    int k = v >> 7;
    int m = v & 127;
    Ash[m * LDA + k] = f2bf(weight[(size_t)k * COUT + d0 + m]);
  }
  if (SEG) {
    for (int v = tid; v < KSEG * 128; v += 256) segsh[v] = 0.0f;
    if (tid < KSEG) cntsh[tid] = 0.0f;
  } else {
    if (tid < 128) ssh[tid] = s[b * COUT + d0 + tid];
  }

  const v4f* x4 = (const v4f*)x;
  const v8f  zacc = {0.f,0.f,0.f,0.f,0.f,0.f,0.f,0.f};
  const int  koff = (lane & 16) ? 8 : 0;
  const int  rlo  = lane & 15;
  const int  hi8  = (lane >> 4) * 8;

  for (int ch = 0; ch < 4; ++ch) {
    const int pixc = pix0 + ch * 128;
    __syncthreads();
    // Stage x tile [k=128][n=128] as Bsh[n][k] (coalesced float4 reads).
    for (int i = 0; i < 16; ++i) {
      int v  = tid + i * 256;
      int k  = v >> 5;
      int n4 = v & 31;
      v4f xv = x4[((size_t)b * CIN + k) * (HW / 4) + (pixc >> 2) + n4];
#pragma unroll
      for (int j = 0; j < 4; ++j)
        Bsh[(n4 * 4 + j) * LDA + k] = f2bf(xv[j]);
    }
    if (tid < 128) idxsh[tid] = index[(size_t)b * HW + pixc + tid] & (KSEG - 1);
    __syncthreads();

    if (SEG && m_tile == 0 && tid < 128)
      atomicAdd(&cntsh[idxsh[tid]], 1.0f);

    v8f acc[2][4];
#pragma unroll
    for (int tm = 0; tm < 2; ++tm)
#pragma unroll
      for (int tn = 0; tn < 4; ++tn) acc[tm][tn] = zacc;

#pragma unroll
    for (int ks = 0; ks < 4; ++ks) {
      const int kk = ks * 32;
      v16bf afrag[2], bfrag[4];
#pragma unroll
      for (int tm = 0; tm < 2; ++tm)
        afrag[tm] = lds_frag(&Ash[(m_off + tm * 16 + rlo) * LDA + kk + koff]);
#pragma unroll
      for (int tn = 0; tn < 4; ++tn)
        bfrag[tn] = lds_frag(&Bsh[(n_off + tn * 16 + rlo) * LDA + kk + koff]);
#pragma unroll
      for (int tm = 0; tm < 2; ++tm)
#pragma unroll
        for (int tn = 0; tn < 4; ++tn)
          acc[tm][tn] = __builtin_amdgcn_wmma_f32_16x16x32_bf16(
              false, afrag[tm], false, bfrag[tn],
              (short)0, acc[tm][tn], false, false);
    }

    if (SEG) {
      // D layout: lane[3:0]=N, VGPR r -> M = hi8 + r. LDS-atomic segment reduce.
#pragma unroll
      for (int tm = 0; tm < 2; ++tm)
#pragma unroll
        for (int tn = 0; tn < 4; ++tn) {
          int nloc = n_off + tn * 16 + rlo;
          int kidx = idxsh[nloc];
#pragma unroll
          for (int r = 0; r < 8; ++r) {
            int mloc = m_off + tm * 16 + hi8 + r;
            atomicAdd(&segsh[kidx * 128 + mloc], acc[tm][tn][r]);
          }
        }
    } else {
#pragma unroll
      for (int tm = 0; tm < 2; ++tm)
#pragma unroll
        for (int tn = 0; tn < 4; ++tn) {
          int nloc = n_off + tn * 16 + rlo;
          int pix  = pixc + nloc;
#pragma unroll
          for (int r = 0; r < 8; ++r) {
            int mloc = m_off + tm * 16 + hi8 + r;
            out[((size_t)b * COUT + d0 + mloc) * HW + pix] =
                acc[tm][tn][r] + ssh[mloc];
          }
        }
    }
  }

  if (SEG) {
    __syncthreads();
    for (int v = tid; v < KSEG * 128; v += 256) {
      int k = v >> 7, m = v & 127;
      atomicAdd(&seg_sum[((size_t)b * KSEG + k) * COUT + d0 + m], segsh[v]);
    }
    if (m_tile == 0 && tid < KSEG)
      atomicAdd(&cnt[b * KSEG + tid], cntsh[tid]);
  }
}

__global__ void zero_kernel(float* p, int n) {
  int i = blockIdx.x * 256 + threadIdx.x;
  if (i < n) p[i] = 0.0f;
}

// M2[r][c] = sum_t Wp[r][t] * Wp[c][t]
__global__ void m2_kernel(const float* __restrict__ Wp, float* __restrict__ M2) {
  __shared__ float row[COUT];
  int r = blockIdx.x, c = threadIdx.x;
  row[c] = Wp[(size_t)r * COUT + c];
  __syncthreads();
  float a = 0.f;
  for (int t = 0; t < COUT; ++t) a += row[t] * Wp[(size_t)c * COUT + t];
  M2[(size_t)r * COUT + c] = a;
}

__global__ void means_kernel(const float* __restrict__ seg,
                             const float* __restrict__ cnt,
                             float* __restrict__ means) {
  int bk = blockIdx.x;                 // b*64 + k
  float cn  = cnt[bk];
  float inv = cn > 0.f ? 1.0f / cn : 0.0f;
  means[(size_t)bk * COUT + threadIdx.x] = seg[(size_t)bk * COUT + threadIdx.x] * inv;
}

// P[bk][d] = sum_c means[bk][c] * M2[c][d]
__global__ void p_kernel(const float* __restrict__ means,
                         const float* __restrict__ M2,
                         float* __restrict__ P) {
  __shared__ float mu[COUT];
  int bk = blockIdx.x, d = threadIdx.x;
  mu[d] = means[(size_t)bk * COUT + d];
  __syncthreads();
  float a = 0.f;
  for (int c = 0; c < COUT; ++c) a += mu[c] * M2[(size_t)c * COUT + d];
  P[(size_t)bk * COUT + d] = a;
}

// G[b][i][j] = sum_d P[b][i][d] * means[b][j][d]
__global__ void g_kernel(const float* __restrict__ P,
                         const float* __restrict__ means,
                         float* __restrict__ G) {
  int b = blockIdx.x;
  int p = blockIdx.y * 256 + threadIdx.x;   // 0..4095
  int i = p >> 6, j = p & 63;
  const float* Pi = P     + ((size_t)b * KSEG + i) * COUT;
  const float* Mj = means + ((size_t)b * KSEG + j) * COUT;
  float a = 0.f;
  for (int d = 0; d < COUT; ++d) a += Pi[d] * Mj[d];
  G[(size_t)b * KSEG * KSEG + p] = a;
}

// w[b][j] = sum_{i!=j} exp(-(G[i][i] + G[j][j] - 2*G[i][j]))
__global__ void w_kernel(const float* __restrict__ G, float* __restrict__ w) {
  int b = blockIdx.x, j = threadIdx.x;
  const float* Gb = G + (size_t)b * KSEG * KSEG;
  float gjj = Gb[j * KSEG + j];
  float a = 0.f;
  for (int i = 0; i < KSEG; ++i) {
    if (i == j) continue;
    a += expf(-(Gb[i * KSEG + i] + gjj - 2.0f * Gb[i * KSEG + j]));
  }
  w[b * KSEG + j] = a;
}

// s[b][d] = sum_j w[b][j] * means[b][j][d]
__global__ void s_kernel(const float* __restrict__ w,
                         const float* __restrict__ means,
                         float* __restrict__ s) {
  int b = blockIdx.x, d = threadIdx.x;
  float a = 0.f;
  for (int j = 0; j < KSEG; ++j)
    a += w[b * KSEG + j] * means[((size_t)b * KSEG + j) * COUT + d];
  s[b * COUT + d] = a;
}

extern "C" void kernel_launch(void* const* d_in, const int* in_sizes, int n_in,
                              void* d_out, int out_size, void* d_ws, size_t ws_size,
                              hipStream_t stream) {
  const float* x      = (const float*)d_in[0];
  const int*   index  = (const int*)d_in[1];
  const float* weight = (const float*)d_in[2];
  const float* Wp     = (const float*)d_in[3];
  float* out = (float*)d_out;

  float* ws    = (float*)d_ws;
  float* seg   = ws;                  // 8*64*256  = 131072
  float* cnt   = seg   + 131072;      // 8*64      = 512
  float* means = cnt   + 512;         // 131072
  float* M2    = means + 131072;      // 65536
  float* P     = M2    + 65536;       // 131072
  float* G     = P     + 131072;      // 8*64*64   = 32768
  float* w     = G     + 32768;       // 512
  float* s     = w     + 512;         // 8*256     = 2048

  // Zero only the atomically-accumulated buffers (seg + cnt) every call.
  zero_kernel<<<(131584 + 255) / 256, 256, 0, stream>>>(ws, 131584);

  dim3 grid(HW / 512, COUT / 128, 8);
  // Pass 1: GEMM + segment sums (x read once, no xp store).
  conv_gemm_kernel<true><<<grid, 256, 0, stream>>>(
      x, index, weight, nullptr, nullptr, seg, cnt);

  // Tiny graph math on the K=64 segment means.
  m2_kernel   <<<COUT, COUT, 0, stream>>>(Wp, M2);
  means_kernel<<<8 * KSEG, COUT, 0, stream>>>(seg, cnt, means);
  p_kernel    <<<8 * KSEG, COUT, 0, stream>>>(means, M2, P);
  g_kernel    <<<dim3(8, 16), 256, 0, stream>>>(P, means, G);
  w_kernel    <<<8, KSEG, 0, stream>>>(G, w);
  s_kernel    <<<8, COUT, 0, stream>>>(w, means, s);

  // Pass 2: recompute GEMM, add broadcast s, store final output.
  conv_gemm_kernel<false><<<grid, 256, 0, stream>>>(
      x, index, weight, s, out, nullptr, nullptr);
}
